// TemporalCrosscoder_16569983828625
// MI455X (gfx1250) — matrix operand order repaired and checked by reading, not compiled, not run on hardware
//
#include <hip/hip_runtime.h>
#include <hip/hip_bf16.h>

// Problem dims
#define D_IN   768
#define D_SAE  16384
#define T_CTX  4
#define K_TOT  (T_CTX * D_IN)   // 3072
#define BATCH  1024
#define KSEL   64

// GEMM tiling
#define BM 128
#define BN 128
#define BK 32
#define LDA 48   // padded LDS row stride in halfs (96B, 16B-aligned)

typedef _Float16 v16h __attribute__((ext_vector_type(16)));
typedef _Float16 v8h  __attribute__((ext_vector_type(8)));
typedef _Float16 v4h  __attribute__((ext_vector_type(4)));
typedef float    v8f  __attribute__((ext_vector_type(8)));

// ---------------------------------------------------------------------------
// Encode: pre[b, n] = relu( sum_k x[b,k] * W[k,n] + b_enc[n] )
// M=1024, N=16384, K=3072. f32 global -> f16 LDS tiles -> v_wmma_f32_16x16x32_f16
// ---------------------------------------------------------------------------
__global__ __launch_bounds__(256)
void encode_gemm(const float* __restrict__ X, const float* __restrict__ W,
                 const float* __restrict__ benc, float* __restrict__ pre) {
    __shared__ _Float16 As[BM * LDA];   // [m][k], 12 KB
    __shared__ _Float16 Bs[BN * LDA];   // [n][k] (transposed), 12 KB

    const int tid   = threadIdx.x;
    const int lane  = tid & 31;
    const int wave  = tid >> 5;
    const int waveM = wave & 3;   // 4 waves along M  -> 32 rows each
    const int waveN = wave >> 2;  // 2 waves along N  -> 64 cols each
    const int m0 = blockIdx.y * BM;
    const int n0 = blockIdx.x * BN;

    v8f acc[2][4];
    {
        v8f z = {};
        #pragma unroll
        for (int i = 0; i < 2; ++i)
            #pragma unroll
            for (int j = 0; j < 4; ++j) acc[i][j] = z;
    }

    // B-tile micro-tile coords: each thread owns a 4(k) x 4(n) patch
    const int bk = (tid >> 5) * 4;   // 0..28
    const int bn = (tid & 31) * 4;   // 0..124

    for (int kt = 0; kt < K_TOT; kt += BK) {
        __syncthreads();

        // ---- stage A tile: 128x32 f32 -> f16, row-major [m][k] ----
        #pragma unroll
        for (int i = 0; i < 4; ++i) {
            int fidx = tid + 256 * i;          // 0..1023
            int m    = fidx & 127;
            int c4   = fidx >> 7;              // 0..7 (float4 column)
            const float4 v = *(const float4*)(X + (size_t)(m0 + m) * K_TOT + kt + c4 * 4);
            v4h h;
            h[0] = (_Float16)v.x; h[1] = (_Float16)v.y;
            h[2] = (_Float16)v.z; h[3] = (_Float16)v.w;
            *(v4h*)(&As[m * LDA + c4 * 4]) = h;
        }

        // ---- stage B tile: 32x128 f32 -> f16 transposed to [n][k] ----
        {
            const float* wp = W + (size_t)(kt + bk) * D_SAE + n0 + bn;
            const float4 r0 = *(const float4*)(wp + 0 * (size_t)D_SAE);
            const float4 r1 = *(const float4*)(wp + 1 * (size_t)D_SAE);
            const float4 r2 = *(const float4*)(wp + 2 * (size_t)D_SAE);
            const float4 r3 = *(const float4*)(wp + 3 * (size_t)D_SAE);
            v4h c0 = { (_Float16)r0.x, (_Float16)r1.x, (_Float16)r2.x, (_Float16)r3.x };
            v4h c1 = { (_Float16)r0.y, (_Float16)r1.y, (_Float16)r2.y, (_Float16)r3.y };
            v4h c2 = { (_Float16)r0.z, (_Float16)r1.z, (_Float16)r2.z, (_Float16)r3.z };
            v4h c3 = { (_Float16)r0.w, (_Float16)r1.w, (_Float16)r2.w, (_Float16)r3.w };
            *(v4h*)(&Bs[(bn + 0) * LDA + bk]) = c0;
            *(v4h*)(&Bs[(bn + 1) * LDA + bk]) = c1;
            *(v4h*)(&Bs[(bn + 2) * LDA + bk]) = c2;
            *(v4h*)(&Bs[(bn + 3) * LDA + bk]) = c3;
        }

        // prefetch next k-tile (emits global_prefetch_b8)
        if (kt + BK < K_TOT) {
            __builtin_prefetch(X + (size_t)(m0 + (tid & 127)) * K_TOT + kt + BK, 0, 3);
            __builtin_prefetch(W + (size_t)(kt + BK + bk) * D_SAE + n0 + bn, 0, 3);
        }

        __syncthreads();

        // ---- fragments ----
        // A 16x32 f16 layout: lanes 0-15 -> K 0..7 / 16..23 ; lanes 16-31 -> K 8..15 / 24..31
        v16h aF[2];
        #pragma unroll
        for (int sm = 0; sm < 2; ++sm) {
            int m_l = waveM * 32 + sm * 16 + (lane & 15);
            int kh  = (lane >> 4) * 8;
            v8h lo = *(const v8h*)(&As[m_l * LDA + kh]);
            v8h hi = *(const v8h*)(&As[m_l * LDA + 16 + kh]);
            aF[sm] = __builtin_shufflevector(lo, hi, 0,1,2,3,4,5,6,7,8,9,10,11,12,13,14,15);
        }
        // B 32x16 f16 layout: lanes 0-15 -> K 0..7 / 16..23 ; lanes 16-31 -> K 8..15 / 24..31 (N = lane&15)
        v16h bF[4];
        #pragma unroll
        for (int sn = 0; sn < 4; ++sn) {
            int n_l = waveN * 64 + sn * 16 + (lane & 15);
            int kb  = (lane >> 4) * 8;
            v8h lo = *(const v8h*)(&Bs[n_l * LDA + kb]);
            v8h hi = *(const v8h*)(&Bs[n_l * LDA + kb + 16]);
            bF[sn] = __builtin_shufflevector(lo, hi, 0,1,2,3,4,5,6,7,8,9,10,11,12,13,14,15);
        }

        #pragma unroll
        for (int sm = 0; sm < 2; ++sm)
            #pragma unroll
            for (int sn = 0; sn < 4; ++sn)
                acc[sm][sn] = __builtin_amdgcn_wmma_f32_16x16x32_f16(
                    false, aF[sm], false, bF[sn], (short)0, acc[sm][sn], false, false);
    }

    // ---- epilogue: bias + relu, store pre ----
    // C/D layout: VGPR r -> M = r + 8*(lane>=16) ; N = lane&15
    #pragma unroll
    for (int sm = 0; sm < 2; ++sm) {
        int mg = m0 + waveM * 32 + sm * 16 + ((lane >> 4) << 3);
        #pragma unroll
        for (int sn = 0; sn < 4; ++sn) {
            int ng = n0 + waveN * 64 + sn * 16 + (lane & 15);
            float bias = benc[ng];
            #pragma unroll
            for (int r = 0; r < 8; ++r) {
                float v = acc[sm][sn][r] + bias;
                v = v > 0.0f ? v : 0.0f;
                pre[(size_t)(mg + r) * D_SAE + ng] = v;
            }
        }
    }
}

// ---------------------------------------------------------------------------
// Top-K (K=64) per row of z (row = 16384, L2-resident). Iterative argmax with
// lowest-index tie-break (matches lax.top_k). Selected entries get -1 sentinel
// (all candidates are >= 0 post-relu), then row is zeroed + topk scattered.
// ---------------------------------------------------------------------------
__global__ __launch_bounds__(256)
void topk_kernel(float* __restrict__ z, float* __restrict__ tv, int* __restrict__ ti) {
    const int row = blockIdx.x;
    const int tid = threadIdx.x;
    float* p = z + (size_t)row * D_SAE;

    __shared__ float rv[256];
    __shared__ int   ri[256];
    __shared__ float topv[KSEL];
    __shared__ int   topi[KSEL];

    for (int sel = 0; sel < KSEL; ++sel) {
        float bm = -2.0f; int bi = 0;
        for (int j = tid; j < D_SAE; j += 256) {
            float v = p[j];
            if (v > bm) { bm = v; bi = j; }   // ascending scan => first (lowest) index kept
        }
        rv[tid] = bm; ri[tid] = bi;
        __syncthreads();
        for (int s = 128; s > 0; s >>= 1) {
            if (tid < s) {
                float ov = rv[tid + s]; int oi = ri[tid + s];
                if (ov > rv[tid] || (ov == rv[tid] && oi < ri[tid])) { rv[tid] = ov; ri[tid] = oi; }
            }
            __syncthreads();
        }
        if (tid == 0) {
            topv[sel] = rv[0]; topi[sel] = ri[0];
            p[ri[0]] = -1.0f;                  // remove from candidates
        }
        __threadfence_block();
        __syncthreads();
    }

    // zero row, then scatter selected values
    float4 zero4 = make_float4(0.0f, 0.0f, 0.0f, 0.0f);
    for (int j = tid; j < D_SAE / 4; j += 256) ((float4*)p)[j] = zero4;
    __threadfence_block();
    __syncthreads();
    if (tid < KSEL) {
        p[topi[tid]] = topv[tid];
        tv[row * KSEL + tid] = topv[tid];
        ti[row * KSEL + tid] = topi[tid];
    }
}

// ---------------------------------------------------------------------------
// Sparse decode: x_hat[b,t,d] = sum_j tv[b,j] * W_dec[t, ti[b,j], d] + b_dec[t,d]
// One block per (b, t); 256 threads cover 768 dims in 3 coalesced strips.
// ---------------------------------------------------------------------------
__global__ __launch_bounds__(256)
void decode_kernel(const float* __restrict__ Wd, const float* __restrict__ bd,
                   const float* __restrict__ tv, const int* __restrict__ ti,
                   float* __restrict__ xhat) {
    const int b   = blockIdx.x;
    const int t   = blockIdx.y;
    const int tid = threadIdx.x;

    __shared__ float sv[KSEL];
    __shared__ int   si[KSEL];
    if (tid < KSEL) { sv[tid] = tv[b * KSEL + tid]; si[tid] = ti[b * KSEL + tid]; }
    __syncthreads();

    const float* bdp = bd + (size_t)t * D_IN;
    float a0 = bdp[tid];
    float a1 = bdp[tid + 256];
    float a2 = bdp[tid + 512];

    #pragma unroll 4
    for (int j = 0; j < KSEL; ++j) {
        float v = sv[j];
        const float* w = Wd + ((size_t)t * D_SAE + si[j]) * D_IN;
        a0 += v * w[tid];
        a1 += v * w[tid + 256];
        a2 += v * w[tid + 512];
    }

    float* o = xhat + ((size_t)b * T_CTX + t) * D_IN;
    o[tid]       = a0;
    o[tid + 256] = a1;
    o[tid + 512] = a2;
}

// ---------------------------------------------------------------------------
extern "C" void kernel_launch(void* const* d_in, const int* in_sizes, int n_in,
                              void* d_out, int out_size, void* d_ws, size_t ws_size,
                              hipStream_t stream) {
    (void)in_sizes; (void)n_in; (void)out_size; (void)ws_size;
    const float* x    = (const float*)d_in[0];   // (B, T, D_IN)
    const float* Wenc = (const float*)d_in[1];   // (T, D_IN, D_SAE) == (K_TOT, D_SAE)
    const float* benc = (const float*)d_in[2];   // (D_SAE)
    const float* Wdec = (const float*)d_in[3];   // (T, D_SAE, D_IN)
    const float* bdec = (const float*)d_in[4];   // (T, D_IN)

    float* xhat = (float*)d_out;                                   // (B, T, D_IN)
    float* z    = (float*)d_out + (size_t)BATCH * T_CTX * D_IN;    // (B, D_SAE)

    float* tv = (float*)d_ws;                                      // (B, 64) values
    int*   ti = (int*)((char*)d_ws + (size_t)BATCH * KSEL * sizeof(float)); // (B, 64) indices

    encode_gemm<<<dim3(D_SAE / BN, BATCH / BM), 256, 0, stream>>>(x, Wenc, benc, z);
    topk_kernel<<<dim3(BATCH), 256, 0, stream>>>(z, tv, ti);
    decode_kernel<<<dim3(BATCH, T_CTX), 256, 0, stream>>>(Wdec, bdec, tv, ti, xhat);
}
